// Generator_72713796321891
// MI455X (gfx1250) — matrix operand order repaired and checked by reading, not compiled, not run on hardware
//
#include <hip/hip_runtime.h>
#include <hip/hip_bf16.h>
#include <cstdint>

typedef __attribute__((ext_vector_type(16))) _Float16 v16h;
typedef __attribute__((ext_vector_type(8)))  _Float16 v8h;
typedef __attribute__((ext_vector_type(8)))  float    v8f;

#define NZ   128
#define H1   512
#define H2   1024
#define IMG  784
#define NGEN 10
#define BATCH 4096
#define MAXT 272   // max 16-row tiles after per-bucket 16-alignment

// ---------------- workspace layout (bytes) ----------------
#define OFF_ZH   ((size_t)0)                                  // 4096*128 f16
#define OFF_W1T  (OFF_ZH  + (size_t)BATCH*NZ*2)               // 10*512*128 f16
#define OFF_W2T  (OFF_W1T + (size_t)NGEN*H1*NZ*2)             // 10*1024*512 f16
#define OFF_W3T  (OFF_W2T + (size_t)NGEN*H2*H1*2)             // 10*784*1024 f16
#define OFF_H1   (OFF_W3T + (size_t)NGEN*IMG*H2*2)            // 4096*512 f16
#define OFF_H2   (OFF_H1  + (size_t)BATCH*H1*2)               // 4096*1024 f16
#define OFF_LIST (OFF_H2  + (size_t)BATCH*H2*2)               // MAXT*16 int
#define OFF_TILG (OFF_LIST + (size_t)MAXT*16*4)               // MAXT int

// ---------------- fp32 -> f16 elementwise ----------------
__global__ __launch_bounds__(256) void cvt_half(const float* __restrict__ in,
                                                _Float16* __restrict__ out, int n) {
  int i = blockIdx.x * blockDim.x + threadIdx.x;
  if (i < n) out[i] = (_Float16)in[i];
}

// ---------------- [G][K][N] fp32 -> [G][N][K] f16 (tiled transpose) ----------------
__global__ __launch_bounds__(256) void transpose_cvt(const float* __restrict__ in,
                                                     _Float16* __restrict__ out,
                                                     int K, int N) {
  __shared__ float t[32][33];
  int g  = blockIdx.z;
  int n0 = blockIdx.x * 32;
  int k0 = blockIdx.y * 32;
  const float* src = in  + (size_t)g * K * N;
  _Float16*    dst = out + (size_t)g * N * K;
#pragma unroll
  for (int r = 0; r < 4; ++r) {
    int k = k0 + threadIdx.y + r * 8;
    int n = n0 + threadIdx.x;
    float v = 0.0f;
    if (k < K && n < N) v = src[(size_t)k * N + n];
    t[threadIdx.y + r * 8][threadIdx.x] = v;
  }
  __syncthreads();
#pragma unroll
  for (int r = 0; r < 4; ++r) {
    int n = n0 + threadIdx.y + r * 8;
    int k = k0 + threadIdx.x;
    if (n < N && k < K) dst[(size_t)n * K + k] = (_Float16)t[threadIdx.x][threadIdx.y + r * 8];
  }
}

// ---------------- bucketize samples by generator (single workgroup) ----------------
__global__ __launch_bounds__(256) void bucketize(const int* __restrict__ gidx,
                                                 int B, int G,
                                                 int* __restrict__ list,
                                                 int* __restrict__ tileG) {
  __shared__ int cnt[16], off[16], pos[16];
  int tid = threadIdx.x;
  if (tid < G) { cnt[tid] = 0; pos[tid] = 0; }
  __syncthreads();
  for (int b = tid; b < B; b += blockDim.x) atomicAdd(&cnt[gidx[b]], 1);
  __syncthreads();
  if (tid == 0) {
    int run = 0;
    for (int g = 0; g < G; ++g) { off[g] = run; run += ((cnt[g] + 15) >> 4) << 4; }
  }
  __syncthreads();
  for (int i = tid; i < MAXT * 16; i += blockDim.x) list[i]  = -1;
  for (int i = tid; i < MAXT;      i += blockDim.x) tileG[i] = -1;
  __syncthreads();
  if (tid == 0) {
    for (int g = 0; g < G; ++g) {
      int t0 = off[g] >> 4, nt = (cnt[g] + 15) >> 4;
      for (int i = 0; i < nt; ++i) tileG[t0 + i] = g;
    }
  }
  for (int b = tid; b < B; b += blockDim.x) {
    int g = gidx[b];
    int p = atomicAdd(&pos[g], 1);
    list[off[g] + p] = b;
  }
}

// ---------------- K-dim WMMA chain over NA column tiles (NA compile-time) ----------------
struct Acc4 { v8f c0, c1, c2, c3; };

template <int K, int NA>
__device__ __forceinline__ Acc4 kchain(const _Float16* As, int aoff,
                                       const _Float16* __restrict__ wbase) {
  Acc4 c = {{}, {}, {}, {}};
  for (int kb = 0; kb < K / 32; ++kb) {
    v8h x0 = *(const v8h*)(As + aoff + kb * 32);
    v8h x1 = *(const v8h*)(As + aoff + kb * 32 + 8);
    v16h a = __builtin_shufflevector(x0, x1,
             0, 1, 2, 3, 4, 5, 6, 7, 8, 9, 10, 11, 12, 13, 14, 15);
    {
      v16h b = *(const v16h*)(wbase + kb * 32);
      c.c0 = __builtin_amdgcn_wmma_f32_16x16x32_f16(false, a, false, b, (short)0, c.c0, false, false);
    }
    if constexpr (NA > 1) {
      v16h b = *(const v16h*)(wbase + (size_t)16 * K + kb * 32);
      c.c1 = __builtin_amdgcn_wmma_f32_16x16x32_f16(false, a, false, b, (short)0, c.c1, false, false);
    }
    if constexpr (NA > 2) {
      v16h b = *(const v16h*)(wbase + (size_t)32 * K + kb * 32);
      c.c2 = __builtin_amdgcn_wmma_f32_16x16x32_f16(false, a, false, b, (short)0, c.c2, false, false);
    }
    if constexpr (NA > 3) {
      v16h b = *(const v16h*)(wbase + (size_t)48 * K + kb * 32);
      c.c3 = __builtin_amdgcn_wmma_f32_16x16x32_f16(false, a, false, b, (short)0, c.c3, false, false);
    }
  }
  return c;
}

// ---------------- epilogue for one 16x16 output tile ----------------
// srow[i]: sample index of output row (i + hl*8), preloaded in registers.
template <int N, int ACT>
__device__ __forceinline__ void epilogue(const v8f& acc, int ct, int l16,
                                         const int (&srow)[8],
                                         const float* __restrict__ biasG,
                                         void* __restrict__ outP) {
  int n = ct * 16 + l16;
  float bn = biasG[n];
#pragma unroll
  for (int i = 0; i < 8; ++i) {
    int s = srow[i];
    if (s >= 0) {
      float v = acc[i] + bn;
      if constexpr (ACT == 0) {
        v = fmaxf(v, 0.0f);
        ((_Float16*)outP)[(size_t)s * N + n] = (_Float16)v;
      } else {
        ((float*)outP)[(size_t)s * N + n] = tanhf(v);
      }
    }
  }
}

// ---------------- WMMA GEMM over one 16-sample tile ----------------
// actIn: [B][K] f16 rows (gathered by sample index); wt: [G][N][K] f16 (transposed)
// ACT==0: relu, f16 out.  ACT==1: tanh, f32 out.
template <int K, int N, int ACT>
__global__ __launch_bounds__(256) void gemm_tile(const _Float16* __restrict__ actIn,
                                                 const _Float16* __restrict__ wt,
                                                 const float* __restrict__ bias,
                                                 const int* __restrict__ list,
                                                 const int* __restrict__ tileG,
                                                 void* __restrict__ outP) {
  static_assert(K % 32 == 0 && N % 16 == 0, "shape");
  constexpr int AS   = K + 8;          // +16B row pad: kills LDS bank conflicts
  constexpr int nct  = N / 16;         // 16-wide column tiles
  constexpr int NPASS = (nct + 31) / 32;
  constexpr bool TAIL = (nct % 32) != 0;
  __shared__ _Float16 As[16 * AS];
  __shared__ int Ss[16];

  const int tile = blockIdx.x;
  const int g = __builtin_amdgcn_readfirstlane(tileG[tile]);
  if (g < 0) return;                   // scalar branch, block-uniform

  const int tid = threadIdx.x;
  if (tid < 16) Ss[tid] = list[tile * 16 + tid];

  // gather A tile (16 rows x K) into LDS, 16-byte chunks
  constexpr int CPR = K / 8;           // 8-half chunks per row
  for (int c = tid; c < 16 * CPR; c += 256) {
    int r = c / CPR, cc = c % CPR;
    int s = list[tile * 16 + r];
    v8h v = {};
    if (s >= 0) v = *(const v8h*)(actIn + (size_t)s * K + (size_t)cc * 8);
    *(v8h*)(As + r * AS + cc * 8) = v;
  }
  __syncthreads();

  const int lane = tid & 31;
  const int wave = __builtin_amdgcn_readfirstlane(tid >> 5);  // SGPR wave id
  const int hl   = lane >> 4;
  const int l16  = lane & 15;
  const int aoff = l16 * AS + hl * 16; // A frag: K {0..15} / {16..31}
  const float* biasG = bias + (size_t)g * N;

  // preload this lane's 8 output-row sample indices into registers
  int srow[8];
#pragma unroll
  for (int i = 0; i < 8; ++i) srow[i] = Ss[i + hl * 8];

#pragma unroll
  for (int p = 0; p < NPASS; ++p) {
    const int ct0 = p * 32 + wave * 4;
    const _Float16* wbase =
        wt + ((size_t)g * N + (size_t)ct0 * 16 + l16) * K + hl * 16;

    if constexpr (TAIL) {
      int na = nct - ct0; if (na > 4) na = 4;
      na = __builtin_amdgcn_readfirstlane(na);   // scalar -> s_cbranch guards
      if (na <= 0) break;
      if (na < 4) {
        if (na == 3) {
          Acc4 c = kchain<K, 3>(As, aoff, wbase);
          epilogue<N, ACT>(c.c0, ct0 + 0, l16, srow, biasG, outP);
          epilogue<N, ACT>(c.c1, ct0 + 1, l16, srow, biasG, outP);
          epilogue<N, ACT>(c.c2, ct0 + 2, l16, srow, biasG, outP);
        } else if (na == 2) {
          Acc4 c = kchain<K, 2>(As, aoff, wbase);
          epilogue<N, ACT>(c.c0, ct0 + 0, l16, srow, biasG, outP);
          epilogue<N, ACT>(c.c1, ct0 + 1, l16, srow, biasG, outP);
        } else {
          Acc4 c = kchain<K, 1>(As, aoff, wbase);
          epilogue<N, ACT>(c.c0, ct0 + 0, l16, srow, biasG, outP);
        }
        continue;
      }
    }
    Acc4 c = kchain<K, 4>(As, aoff, wbase);
    epilogue<N, ACT>(c.c0, ct0 + 0, l16, srow, biasG, outP);
    epilogue<N, ACT>(c.c1, ct0 + 1, l16, srow, biasG, outP);
    epilogue<N, ACT>(c.c2, ct0 + 2, l16, srow, biasG, outP);
    epilogue<N, ACT>(c.c3, ct0 + 3, l16, srow, biasG, outP);
  }
}

extern "C" void kernel_launch(void* const* d_in, const int* in_sizes, int n_in,
                              void* d_out, int out_size, void* d_ws, size_t ws_size,
                              hipStream_t stream) {
  const float* z    = (const float*)d_in[0];
  const int*   gidx = (const int*)d_in[1];
  const float* W1   = (const float*)d_in[2];
  const float* b1   = (const float*)d_in[3];
  const float* W2   = (const float*)d_in[4];
  const float* b2   = (const float*)d_in[5];
  const float* W3   = (const float*)d_in[6];
  const float* b3   = (const float*)d_in[7];
  float* out = (float*)d_out;

  char* ws = (char*)d_ws;
  _Float16* zh  = (_Float16*)(ws + OFF_ZH);
  _Float16* w1t = (_Float16*)(ws + OFF_W1T);
  _Float16* w2t = (_Float16*)(ws + OFF_W2T);
  _Float16* w3t = (_Float16*)(ws + OFF_W3T);
  _Float16* h1  = (_Float16*)(ws + OFF_H1);
  _Float16* h2  = (_Float16*)(ws + OFF_H2);
  int* list  = (int*)(ws + OFF_LIST);
  int* tileG = (int*)(ws + OFF_TILG);

  // 1) convert z to f16
  {
    int n = BATCH * NZ;
    cvt_half<<<(n + 255) / 256, 256, 0, stream>>>(z, zh, n);
  }
  // 2) transpose-convert weights to [G][N][K] f16
  {
    dim3 blk(32, 8);
    transpose_cvt<<<dim3(H1 / 32,  NZ / 32, NGEN), blk, 0, stream>>>(W1, w1t, NZ, H1);
    transpose_cvt<<<dim3(H2 / 32,  H1 / 32, NGEN), blk, 0, stream>>>(W2, w2t, H1, H2);
    transpose_cvt<<<dim3((IMG + 31) / 32, H2 / 32, NGEN), blk, 0, stream>>>(W3, w3t, H2, IMG);
  }
  // 3) bucket samples by generator
  bucketize<<<1, 256, 0, stream>>>(gidx, BATCH, NGEN, list, tileG);

  // 4) three WMMA GEMM layers (selected branch only)
  gemm_tile<NZ, H1, 0><<<MAXT, 256, 0, stream>>>(zh, w1t, b1, list, tileG, (void*)h1);
  gemm_tile<H1, H2, 0><<<MAXT, 256, 0, stream>>>(h1, w2t, b2, list, tileG, (void*)h2);
  gemm_tile<H2, IMG, 1><<<MAXT, 256, 0, stream>>>(h2, w3t, b3, list, tileG, (void*)out);
}
